// HeatwaveGNN_55800215109810
// MI455X (gfx1250) — matrix operand order repaired and checked by reading, not compiled, not run on hardware
//
#include <hip/hip_runtime.h>
#include <hip/hip_bf16.h>
#include <math.h>

typedef __attribute__((ext_vector_type(16))) __bf16 v16bf;
typedef __attribute__((ext_vector_type(8)))  __bf16 v8bf;
typedef __attribute__((ext_vector_type(8)))  float  v8f;

#define N_NODES 50000
#define N_EDGES 640000
#define F_IN    128
#define HID     256

// ---------------- degree / normalization ----------------
__global__ void k_init_deg(float* deg, int n) {
    int i = blockIdx.x * blockDim.x + threadIdx.x;
    if (i < n) deg[i] = 1.0f;  // self loop contributes 1
}

__global__ void k_count_deg(const int* __restrict__ ei, float* deg, int e) {
    int i = blockIdx.x * blockDim.x + threadIdx.x;
    if (i < e) {
        int d = ei[e + i];  // edge_index[1][i]
        atomicAdd(&deg[d], 1.0f);
    }
}

__global__ void k_rsqrt(float* deg, int n) {
    int i = blockIdx.x * blockDim.x + threadIdx.x;
    if (i < n) deg[i] = rsqrtf(deg[i]);  // deg >= 1 always (self loops)
}

// ---------------- conversions ----------------
__global__ void k_f32_to_bf16(const float* __restrict__ in, __bf16* __restrict__ out, int n) {
    int i = blockIdx.x * blockDim.x + threadIdx.x;
    if (i < n) out[i] = (__bf16)in[i];
}

// WT[n*K + k] = (bf16) W[k*N + n]   (transpose so B-frags load contiguously)
__global__ void k_w_transpose_bf16(const float* __restrict__ W, __bf16* __restrict__ WT,
                                   int K, int N) {
    int i = blockIdx.x * blockDim.x + threadIdx.x;
    if (i < K * N) {
        int k = i / N, n = i % N;
        WT[n * K + k] = (__bf16)W[i];
    }
}

// ---------------- bf16 WMMA GEMM: H[M,N] = A[M,K] * W[K,N] (W given transposed) ----
// one wave per 16x16 output tile; K stepped by 32 (v_wmma_f32_16x16x32_bf16)
__global__ void k_gemm_wmma(const __bf16* __restrict__ A, const __bf16* __restrict__ WT,
                            float* __restrict__ H, int M, int N, int K) {
    int wave = (int)((blockIdx.x * blockDim.x + threadIdx.x) >> 5);
    int lane = threadIdx.x & 31;
    int ntiles = N >> 4;
    int total = (M >> 4) * ntiles;
    if (wave >= total) return;           // uniform per wave -> EXEC stays all-1s
    int rt = wave / ntiles;
    int ct = wave - rt * ntiles;
    int r    = lane & 15;                // row within A-tile / col within B-tile
    int half = lane >> 4;

    const __bf16* arow = A  + (long long)(rt * 16 + r) * K;
    const __bf16* bcol = WT + (long long)(ct * 16 + r) * K;

    v8f acc = {};
    for (int k = 0; k < K; k += 32) {
        // A 16x32 bf16 frag: lane(r,half): elems j = K[k + half*16 + j]
        v8bf a0 = *(const v8bf*)(arow + k + half * 16);
        v8bf a1 = *(const v8bf*)(arow + k + half * 16 + 8);
        v16bf a;
#pragma unroll
        for (int i = 0; i < 8; ++i) { a[i] = a0[i]; a[i + 8] = a1[i]; }
        // B 32x16 bf16 frag: lane(c,half): elems j = B[k + half*16 + j][c] -> contiguous in WT
        v16bf b = *(const v16bf*)(bcol + k + half * 16);
        acc = __builtin_amdgcn_wmma_f32_16x16x32_bf16(false, a, false, b,
                                                      (short)0, acc, false, false);
    }
    // C/D layout: VGPR v, lanes 0-15 -> M=v, lanes 16-31 -> M=8+v; N = lane&15
    float* out = H + (long long)(rt * 16 + half * 8) * N + ct * 16 + r;
#pragma unroll
    for (int v = 0; v < 8; ++v) out[(long long)v * N] = acc[v];
}

// ---------------- aggregation ----------------
// self-loop term: A[n,:] = H[n,:] * dis[n]^2   (float4 granularity)
__global__ void k_self_init4(const float4* __restrict__ H, const float* __restrict__ dis,
                             float4* __restrict__ A, int n, int Fq) {
    int i = blockIdx.x * blockDim.x + threadIdx.x;
    if (i < n * Fq) {
        int node = i / Fq;
        float d = dis[node]; d = d * d;
        float4 h = H[i];
        A[i] = make_float4(h.x * d, h.y * d, h.z * d, h.w * d);
    }
}

// edge scatter: A[dst,:] += H[src,:] * dis[src]*dis[dst]; 64 lanes per edge (F=256)
__global__ void k_scatter(const int* __restrict__ ei, const float* __restrict__ dis,
                          const float* __restrict__ H, float* A, int E, int F) {
    int t = blockIdx.x * blockDim.x + threadIdx.x;
    int lpe = F >> 2;                  // float4 lanes per edge
    int e = t / lpe;
    int f4 = t - e * lpe;
    if (e >= E) return;
    int s = ei[e];
    int d = ei[E + e];
    float norm = dis[s] * dis[d];
    float4 h = *(const float4*)(H + (long long)s * F + f4 * 4);
    float* ap = A + (long long)d * F + f4 * 4;
    atomicAdd(ap + 0, h.x * norm);
    atomicAdd(ap + 1, h.y * norm);
    atomicAdd(ap + 2, h.z * norm);
    atomicAdd(ap + 3, h.w * norm);
}

// A = relu(A + b); also emit bf16 copy for the next GEMM
__global__ void k_bias_relu_conv(float* A, const float* __restrict__ b,
                                 __bf16* __restrict__ Bb, int n, int F) {
    int i = blockIdx.x * blockDim.x + threadIdx.x;
    if (i < n * F) {
        int f = i % F;
        float v = A[i] + b[f];
        v = v > 0.0f ? v : 0.0f;
        A[i] = v;
        Bb[i] = (__bf16)v;
    }
}

// ---------------- layer 3 (out dim 1) ----------------
__global__ void k_gemv(const float* __restrict__ A, const float* __restrict__ W3,
                       float* __restrict__ z, int n, int F) {
    int wave = (int)((blockIdx.x * blockDim.x + threadIdx.x) >> 5);
    int lane = threadIdx.x & 31;
    if (wave >= n) return;
    const float* row = A + (long long)wave * F;
    float s = 0.0f;
    for (int k = lane; k < F; k += 32) s += row[k] * W3[k];
#pragma unroll
    for (int off = 16; off > 0; off >>= 1) s += __shfl_xor(s, off, 32);
    if (lane == 0) z[wave] = s;
}

__global__ void k_self_init1(const float* __restrict__ z, const float* __restrict__ dis,
                             float* O, int n) {
    int i = blockIdx.x * blockDim.x + threadIdx.x;
    if (i < n) { float d = dis[i]; O[i] = z[i] * d * d; }
}

__global__ void k_scatter1(const int* __restrict__ ei, const float* __restrict__ dis,
                           const float* __restrict__ z, float* O, int E) {
    int e = blockIdx.x * blockDim.x + threadIdx.x;
    if (e < E) {
        int s = ei[e];
        int d = ei[E + e];
        atomicAdd(&O[d], z[s] * dis[s] * dis[d]);
    }
}

__global__ void k_sigmoid(const float* __restrict__ O, const float* __restrict__ b3,
                          float* __restrict__ out, int n) {
    int i = blockIdx.x * blockDim.x + threadIdx.x;
    if (i < n) out[i] = 1.0f / (1.0f + __expf(-(O[i] + b3[0])));
}

// ---------------- orchestration ----------------
extern "C" void kernel_launch(void* const* d_in, const int* in_sizes, int n_in,
                              void* d_out, int out_size, void* d_ws, size_t ws_size,
                              hipStream_t stream) {
    const float* x   = (const float*)d_in[0];
    const int*   ei  = (const int*)d_in[1];   // [2, E] int32 (JAX default x64-off)
    const float* W1  = (const float*)d_in[2];
    const float* b1  = (const float*)d_in[3];
    const float* W2  = (const float*)d_in[4];
    const float* b2  = (const float*)d_in[5];
    const float* W3  = (const float*)d_in[6];
    const float* b3  = (const float*)d_in[7];
    float* out = (float*)d_out;

    char* ws = (char*)d_ws;
    // workspace layout (256B aligned)
    float*  dis  = (float*)(ws + 0);                 // 50000 f32
    float*  z    = (float*)(ws + 200704);            // 50000 f32
    float*  O    = (float*)(ws + 401408);            // 50000 f32
    __bf16* W1T  = (__bf16*)(ws + 602112);           // 256*128 bf16
    __bf16* W2T  = (__bf16*)(ws + 667648);           // 256*256 bf16
    __bf16* Bbf  = (__bf16*)(ws + 798720);           // 50000*256 bf16 (GEMM input)
    float*  H    = (float*)(ws + 26398720);          // 50000*256 f32 (GEMM output)
    float*  A    = (float*)(ws + 77598720);          // 50000*256 f32 (aggregated)
    // total ~128.8 MB

    const int TB = 256;
    // normalization
    k_init_deg <<<(N_NODES + TB - 1) / TB, TB, 0, stream>>>(dis, N_NODES);
    k_count_deg<<<(N_EDGES + TB - 1) / TB, TB, 0, stream>>>(ei, dis, N_EDGES);
    k_rsqrt    <<<(N_NODES + TB - 1) / TB, TB, 0, stream>>>(dis, N_NODES);

    // weights (transposed bf16)
    k_w_transpose_bf16<<<(F_IN * HID + TB - 1) / TB, TB, 0, stream>>>(W1, W1T, F_IN, HID);
    k_w_transpose_bf16<<<(HID * HID + TB - 1) / TB, TB, 0, stream>>>(W2, W2T, HID, HID);

    // ---- layer 1 ----
    k_f32_to_bf16<<<(N_NODES * F_IN + TB - 1) / TB, TB, 0, stream>>>(x, Bbf, N_NODES * F_IN);
    // 3125*16 = 50000 tiles, 8 waves/block -> 6250 full blocks (EXEC all-1s for WMMA)
    k_gemm_wmma<<<6250, TB, 0, stream>>>(Bbf, W1T, H, N_NODES, HID, F_IN);
    k_self_init4<<<(N_NODES * (HID / 4) + TB - 1) / TB, TB, 0, stream>>>(
        (const float4*)H, dis, (float4*)A, N_NODES, HID / 4);
    k_scatter<<<(N_EDGES * (HID / 4) + TB - 1) / TB, TB, 0, stream>>>(ei, dis, H, A, N_EDGES, HID);
    k_bias_relu_conv<<<(N_NODES * HID + TB - 1) / TB, TB, 0, stream>>>(A, b1, Bbf, N_NODES, HID);

    // ---- layer 2 ----
    k_gemm_wmma<<<6250, TB, 0, stream>>>(Bbf, W2T, H, N_NODES, HID, HID);
    k_self_init4<<<(N_NODES * (HID / 4) + TB - 1) / TB, TB, 0, stream>>>(
        (const float4*)H, dis, (float4*)A, N_NODES, HID / 4);
    k_scatter<<<(N_EDGES * (HID / 4) + TB - 1) / TB, TB, 0, stream>>>(ei, dis, H, A, N_EDGES, HID);
    k_bias_relu_conv<<<(N_NODES * HID + TB - 1) / TB, TB, 0, stream>>>(A, b2, Bbf, N_NODES, HID);

    // ---- layer 3 (dim 1) ----
    k_gemv<<<(N_NODES * 32 + TB - 1) / TB, TB, 0, stream>>>(A, W3, z, N_NODES, HID);
    k_self_init1<<<(N_NODES + TB - 1) / TB, TB, 0, stream>>>(z, dis, O, N_NODES);
    k_scatter1<<<(N_EDGES + TB - 1) / TB, TB, 0, stream>>>(ei, dis, z, O, N_EDGES);
    k_sigmoid<<<(N_NODES + TB - 1) / TB, TB, 0, stream>>>(O, b3, out, N_NODES);
}
